// Attention_53352083751420
// MI455X (gfx1250) — matrix operand order repaired and checked by reading, compiled, not run on hardware
//
#include <hip/hip_runtime.h>
#include <hip/hip_bf16.h>
#include <cstdint>

// ---------------------------------------------------------------------------
// Causal multi-head self-attention forward for MI455X (gfx1250, wave32, WMMA)
// B=4, S=1024, E=1024, H=16, HD=64
// All GEMMs use v_wmma_f32_16x16x32_f16. LDS tiles are kept in FRAGMENT-MAJOR
// order ([frag][lane][16 halves]): staging writes are 16B ds_store_b128 (or
// GLOBAL_LOAD_ASYNC_TO_LDS_B128 when the toolchain exposes it), fragment
// reads are 2x ds_load_b128 feeding the WMMA directly.
// ---------------------------------------------------------------------------

typedef __attribute__((ext_vector_type(16))) _Float16 v16h;
typedef __attribute__((ext_vector_type(8)))  _Float16 v8h;
typedef __attribute__((ext_vector_type(8)))  float    v8f;
typedef int v4i __attribute__((__vector_size__(16)));   // matches builtin param

constexpr int B_  = 4;
constexpr int S_  = 1024;
constexpr int E_  = 1024;
constexpr int H_  = 16;
constexpr int HD_ = 64;

#define WMMA_F16(a, b, c) \
    __builtin_amdgcn_wmma_f32_16x16x32_f16(false, (a), false, (b), (short)0, (c), false, false)

// ---------------- async-to-LDS feature probe (gfx1250 ASYNCcnt path) --------
#if defined(__has_builtin)
#if __has_builtin(__builtin_amdgcn_global_load_async_to_lds_b128)
#define HAVE_ASYNC_LDS 1
#endif
#endif

#define GLOBAL_AS __attribute__((address_space(1)))
#define LDS_AS    __attribute__((address_space(3)))

__device__ __forceinline__ void async_copy16(const _Float16* g, _Float16* l) {
#ifdef HAVE_ASYNC_LDS
    __builtin_amdgcn_global_load_async_to_lds_b128(
        (GLOBAL_AS v4i*)(_Float16*)g, (LDS_AS v4i*)l, 0, 0);
#else
    *(v8h*)l = *(const v8h*)g;
#endif
}

__device__ __forceinline__ void wait_async() {
#ifdef HAVE_ASYNC_LDS
#if __has_builtin(__builtin_amdgcn_s_wait_asynccnt)
    __builtin_amdgcn_s_wait_asynccnt(0);
#else
    asm volatile("s_wait_asynccnt 0" ::: "memory");
#endif
#endif
}

// ---------------------------------------------------------------------------
// Staging helpers: move a 64x32 (A: MxK) or 32x64 (B: KxN, 4 col-tiles) panel
// into fragment-major LDS. 128 threads, 256 octets of 8 halves each.
// Source is contiguous along K for every operand (V is stored transposed).
// ---------------------------------------------------------------------------

__device__ __forceinline__ v8h cvt8(const float* g) {
    const float4 f0 = ((const float4*)g)[0];
    const float4 f1 = ((const float4*)g)[1];
    v8h h = { (_Float16)f0.x, (_Float16)f0.y, (_Float16)f0.z, (_Float16)f0.w,
              (_Float16)f1.x, (_Float16)f1.y, (_Float16)f1.z, (_Float16)f1.w };
    return h;
}

// A panel: rows m=0..63 (4 wave sub-tiles of 16), K octets 0..3.
// frag slot: lane = ((oct&1)<<4)+(m&15), ebase = (oct>>1)*8, frag = m>>4
__device__ __forceinline__ void stage_a_f32(_Float16* As, const float* src,
                                            size_t ld, int row0, int k0, int tid) {
#pragma unroll
    for (int j = 0; j < 2; ++j) {
        const int i = tid + j * 128;          // 0..255
        const int m = i >> 2, oct = i & 3;
        const v8h h = cvt8(src + (size_t)(row0 + m) * ld + k0 + oct * 8);
        const int lane  = ((oct & 1) << 4) + (m & 15);
        const int ebase = (oct >> 1) * 8;
        *(v8h*)(As + (((m >> 4) * 32 + lane) * 16 + ebase)) = h;
    }
}

__device__ __forceinline__ void stage_a_f16(_Float16* As, const _Float16* src,
                                            size_t ld, int row0, int k0, int tid) {
#pragma unroll
    for (int j = 0; j < 2; ++j) {
        const int i = tid + j * 128;
        const int m = i >> 2, oct = i & 3;
        const _Float16* g = src + (size_t)(row0 + m) * ld + k0 + oct * 8;
        const int lane  = ((oct & 1) << 4) + (m & 15);
        const int ebase = (oct >> 1) * 8;
        async_copy16(g, As + (((m >> 4) * 32 + lane) * 16 + ebase));
    }
}

// B panel: cols n64=0..63 (4 col-tiles of 16), K octets 0..3; source row n64
// is contiguous along K. frag slot: lane = ((oct&1)<<4)+(n64&15),
// ebase = (oct>>1)*8, frag = n64>>4
__device__ __forceinline__ void stage_b_f32(_Float16* Bs, const float* src,
                                            size_t ld, int col0, int k0, int tid) {
#pragma unroll
    for (int j = 0; j < 2; ++j) {
        const int i = tid + j * 128;
        const int n64 = i >> 2, oct = i & 3;
        const v8h h = cvt8(src + (size_t)(col0 + n64) * ld + k0 + oct * 8);
        const int lane  = ((oct & 1) << 4) + (n64 & 15);
        const int ebase = (oct >> 1) * 8;
        *(v8h*)(Bs + (((n64 >> 4) * 32 + lane) * 16 + ebase)) = h;
    }
}

__device__ __forceinline__ void stage_b_f16(_Float16* Bs, const _Float16* src,
                                            size_t ld, int col0, int k0, int tid) {
#pragma unroll
    for (int j = 0; j < 2; ++j) {
        const int i = tid + j * 128;
        const int n64 = i >> 2, oct = i & 3;
        const _Float16* g = src + (size_t)(col0 + n64) * ld + k0 + oct * 8;
        const int lane  = ((oct & 1) << 4) + (n64 & 15);
        const int ebase = (oct >> 1) * 8;
        async_copy16(g, Bs + (((n64 >> 4) * 32 + lane) * 16 + ebase));
    }
}

// Fragment fetch: one contiguous 32-byte LDS read per operand.
__device__ __forceinline__ v16h load_frag(const _Float16* tile, int frag, int lane) {
    return *(const v16h*)(tile + (frag * 32 + lane) * 16);
}

// ---------------------------------------------------------------------------
// Kernel 1: fused QKV projection. C[r,f] = sum_e X[r,e]*W[f,e] + bias[f]
// Q,K -> [B,H,S,HD] f16 ; V -> TRANSPOSED [B,H,HD,S] f16 (K-contiguous for PV)
// ---------------------------------------------------------------------------
__global__ __launch_bounds__(128) void attn_qkv_gemm(
    const float* __restrict__ X, const float* __restrict__ W,
    const float* __restrict__ bias,
    _Float16* __restrict__ Qw, _Float16* __restrict__ Kw, _Float16* __restrict__ Vt) {
    __shared__ _Float16 As[4 * 32 * 16];
    __shared__ _Float16 Bs[4 * 32 * 16];

    const int row0 = blockIdx.x * 64;            // token rows (B*S = 4096)
    const int col0 = blockIdx.y * 64;            // qkv feature cols (3E = 3072)
    const int tid  = threadIdx.x;
    const int lane = tid & 31;
    const int wave = tid >> 5;

    v8f acc[4] = {};

    for (int k0 = 0; k0 < E_; k0 += 32) {
        stage_a_f32(As, X, E_, row0, k0, tid);
        stage_b_f32(Bs, W, E_, col0, k0, tid);
        if (k0 + 32 < E_) {                      // global_prefetch_b8
            __builtin_prefetch(&X[(size_t)(row0 + (tid >> 1)) * E_ + k0 + 32], 0, 0);
            __builtin_prefetch(&W[(size_t)(col0 + (tid >> 1)) * E_ + k0 + 32], 0, 0);
        }
        __syncthreads();
        const v16h a = load_frag(As, wave, lane);
#pragma unroll
        for (int t = 0; t < 4; ++t)
            acc[t] = WMMA_F16(a, load_frag(Bs, t, lane), acc[t]);
        __syncthreads();
    }

    const int n = lane & 15;
#pragma unroll
    for (int t = 0; t < 4; ++t) {
        const int col = col0 + t * 16 + n;       // [0, 3E)
        const float bv = bias[col];
        const int sec = col >> 10;               // 0=Q 1=K 2=V
        const int cf  = col & (E_ - 1);
        const int hh  = cf >> 6;
        const int d   = cf & (HD_ - 1);
#pragma unroll
        for (int e = 0; e < 8; ++e) {
            const int m = row0 + wave * 16 + (lane >> 4) * 8 + e;  // token row
            const int b = m >> 10;
            const int s = m & (S_ - 1);
            const _Float16 val = (_Float16)(acc[t][e] + bv);
            if (sec == 2) {  // V transposed: [B,H,HD,S]
                Vt[((((size_t)b * H_ + hh) * HD_ + d) * S_) + s] = val;
            } else {
                _Float16* dst = (sec == 0) ? Qw : Kw;
                dst[((((size_t)b * H_ + hh) * S_ + s) * HD_) + d] = val;
            }
        }
    }
}

// ---------------------------------------------------------------------------
// Kernel 2: scores = scale * Q K^T per (b,h). fp32 into att region of d_out.
// Fully causally-masked 64x64 blocks skipped. grid=(S/64,S/64,B*H)
// Q/K tiles are staged via async-to-LDS (pure f16 swizzled copies).
// ---------------------------------------------------------------------------
__global__ __launch_bounds__(128) void attn_score_gemm(
    const _Float16* __restrict__ Qw, const _Float16* __restrict__ Kw,
    float* __restrict__ att) {
    if (blockIdx.y > blockIdx.x) return;

    __shared__ _Float16 As[4 * 32 * 16];
    __shared__ _Float16 Bs[4 * 32 * 16];

    const int row0 = blockIdx.x * 64;
    const int col0 = blockIdx.y * 64;
    const int bh   = blockIdx.z;
    const int tid  = threadIdx.x;
    const int lane = tid & 31;
    const int wave = tid >> 5;

    const _Float16* Qp = Qw + (size_t)bh * S_ * HD_;
    const _Float16* Kp = Kw + (size_t)bh * S_ * HD_;

    v8f acc[4] = {};

#pragma unroll
    for (int k0 = 0; k0 < HD_; k0 += 32) {
        stage_a_f16(As, Qp, HD_, row0, k0, tid);
        stage_b_f16(Bs, Kp, HD_, col0, k0, tid);
        wait_async();
        __syncthreads();
        const v16h a = load_frag(As, wave, lane);
#pragma unroll
        for (int t = 0; t < 4; ++t)
            acc[t] = WMMA_F16(a, load_frag(Bs, t, lane), acc[t]);
        __syncthreads();
    }

    const float scale = 0.125f;                  // HD^-0.5
    const int n = lane & 15;
    float* dst = att + (size_t)bh * S_ * S_;
#pragma unroll
    for (int t = 0; t < 4; ++t) {
#pragma unroll
        for (int e = 0; e < 8; ++e) {
            const int m = row0 + wave * 16 + (lane >> 4) * 8 + e;
            dst[(size_t)m * S_ + col0 + t * 16 + n] = acc[t][e] * scale;
        }
    }
}

// ---------------------------------------------------------------------------
// Kernel 3: in-place causal softmax over each att row. grid=B*H*S, block=256
// ---------------------------------------------------------------------------
__global__ __launch_bounds__(256) void attn_softmax_rows(float* __restrict__ att) {
    __shared__ float sd[256];
    const int row = blockIdx.x;
    const int qr  = row & (S_ - 1);
    const int lim = qr + 1;
    float* p = att + (size_t)row * S_;
    const int tid = threadIdx.x;

    float mx = -__builtin_inff();
    for (int c = tid; c < lim; c += 256) mx = fmaxf(mx, p[c]);
    sd[tid] = mx;
    __syncthreads();
    for (int s = 128; s > 0; s >>= 1) {
        if (tid < s) sd[tid] = fmaxf(sd[tid], sd[tid + s]);
        __syncthreads();
    }
    mx = sd[0];
    __syncthreads();

    float sum = 0.0f;
    for (int c = tid; c < lim; c += 256) sum += __expf(p[c] - mx);
    sd[tid] = sum;
    __syncthreads();
    for (int s = 128; s > 0; s >>= 1) {
        if (tid < s) sd[tid] += sd[tid + s];
        __syncthreads();
    }
    const float inv = 1.0f / sd[0];

    for (int c = tid; c < S_; c += 256) {
        p[c] = (c < lim) ? __expf(p[c] - mx) * inv : 0.0f;
    }
}

// ---------------------------------------------------------------------------
// Kernel 4: context = att @ V per (b,h). A: att fp32->f16 (VALU convert);
// B: V^T rows K-contiguous, staged async. grid=(S/64, B*H). f16 ctx [B,S,E].
// ---------------------------------------------------------------------------
__global__ __launch_bounds__(128) void attn_pv_gemm(
    const float* __restrict__ att, const _Float16* __restrict__ Vt,
    _Float16* __restrict__ ctx) {
    __shared__ _Float16 As[4 * 32 * 16];
    __shared__ _Float16 Bs[4 * 32 * 16];

    const int row0 = blockIdx.x * 64;
    const int bh   = blockIdx.y;
    const int bb   = bh >> 4;
    const int hh   = bh & 15;
    const int tid  = threadIdx.x;
    const int lane = tid & 31;
    const int wave = tid >> 5;

    const float*    Ap = att + (size_t)bh * S_ * S_;
    const _Float16* Vp = Vt  + (size_t)bh * HD_ * S_;   // [HD][S]

    v8f acc[4] = {};

    for (int k0 = 0; k0 < S_; k0 += 32) {
        stage_b_f16(Bs, Vp, S_, 0, k0, tid);            // async (n64 = dim d)
        stage_a_f32(As, Ap, S_, row0, k0, tid);
        if (k0 + 32 < S_)
            __builtin_prefetch(&Ap[(size_t)(row0 + (tid >> 1)) * S_ + k0 + 32], 0, 0);
        wait_async();
        __syncthreads();
        const v16h a = load_frag(As, wave, lane);
#pragma unroll
        for (int t = 0; t < 4; ++t)
            acc[t] = WMMA_F16(a, load_frag(Bs, t, lane), acc[t]);
        __syncthreads();
    }

    const int n = lane & 15;
#pragma unroll
    for (int t = 0; t < 4; ++t) {
        const int col = hh * HD_ + t * 16 + n;
#pragma unroll
        for (int e = 0; e < 8; ++e) {
            const int s = row0 + wave * 16 + (lane >> 4) * 8 + e;
            ctx[((size_t)bb * S_ + s) * E_ + col] = (_Float16)acc[t][e];
        }
    }
}

// ---------------------------------------------------------------------------
// Kernel 5: output projection. out[r,f] = sum_e ctx[r,e]*Wo[f,e] + bo[f]
// grid=(4096/64, 1024/64). ctx staged async; Wo converted fp32->f16.
// ---------------------------------------------------------------------------
__global__ __launch_bounds__(128) void attn_out_gemm(
    const _Float16* __restrict__ ctx, const float* __restrict__ Wo,
    const float* __restrict__ bo, float* __restrict__ out) {
    __shared__ _Float16 As[4 * 32 * 16];
    __shared__ _Float16 Bs[4 * 32 * 16];

    const int row0 = blockIdx.x * 64;
    const int col0 = blockIdx.y * 64;
    const int tid  = threadIdx.x;
    const int lane = tid & 31;
    const int wave = tid >> 5;

    v8f acc[4] = {};

    for (int k0 = 0; k0 < E_; k0 += 32) {
        stage_a_f16(As, ctx, E_, row0, k0, tid);        // async
        stage_b_f32(Bs, Wo, E_, col0, k0, tid);
        if (k0 + 32 < E_)
            __builtin_prefetch(&Wo[(size_t)(col0 + (tid >> 1)) * E_ + k0 + 32], 0, 0);
        wait_async();
        __syncthreads();
        const v16h a = load_frag(As, wave, lane);
#pragma unroll
        for (int t = 0; t < 4; ++t)
            acc[t] = WMMA_F16(a, load_frag(Bs, t, lane), acc[t]);
        __syncthreads();
    }

    const int n = lane & 15;
#pragma unroll
    for (int t = 0; t < 4; ++t) {
        const int col = col0 + t * 16 + n;
        const float bv = bo[col];
#pragma unroll
        for (int e = 0; e < 8; ++e) {
            const int m = row0 + wave * 16 + (lane >> 4) * 8 + e;
            out[(size_t)m * E_ + col] = acc[t][e] + bv;
        }
    }
}

// ---------------------------------------------------------------------------
extern "C" void kernel_launch(void* const* d_in, const int* in_sizes, int n_in,
                              void* d_out, int out_size, void* d_ws, size_t ws_size,
                              hipStream_t stream) {
    const float* x    = (const float*)d_in[0];
    // d_in[1] = mask (causal, applied analytically — unused)
    const float* w_in = (const float*)d_in[2];
    const float* b_in = (const float*)d_in[3];
    const float* w_o  = (const float*)d_in[4];
    const float* b_o  = (const float*)d_in[5];

    float* out = (float*)d_out;                              // [B,S,E]
    float* att = out + (size_t)B_ * S_ * E_;                 // [B*H,S,S]

    const size_t NHEAD = (size_t)B_ * H_ * S_ * HD_;         // 4,194,304 elems
    _Float16* qw  = (_Float16*)d_ws;                         // [B,H,S,HD]
    _Float16* kw  = qw + NHEAD;                              // [B,H,S,HD]
    _Float16* vt  = kw + NHEAD;                              // [B,H,HD,S]
    _Float16* ctx = vt + NHEAD;                              // [B,S,E]
    (void)in_sizes; (void)n_in; (void)out_size; (void)ws_size;

    attn_qkv_gemm<<<dim3(64, 48), 128, 0, stream>>>(x, w_in, b_in, qw, kw, vt);
    attn_score_gemm<<<dim3(S_ / 64, S_ / 64, B_ * H_), 128, 0, stream>>>(qw, kw, att);
    attn_softmax_rows<<<B_ * H_ * S_, 256, 0, stream>>>(att);
    attn_pv_gemm<<<dim3(S_ / 64, B_ * H_), 128, 0, stream>>>(att, vt, ctx);
    attn_out_gemm<<<dim3(64, 16), 128, 0, stream>>>(ctx, w_o, b_o, out);
}
